// ConditionalMixturePrior_4269197492641
// MI455X (gfx1250) — compile-verified, matrix operand order checked by reading, not run on hardware
//
#include <hip/hip_runtime.h>
#include <hip/hip_bf16.h>
#include <math.h>

// ---------------------------------------------------------------------------
// ConditionalMixturePrior GNN for gfx1250 (MI455X), wave32 + WMMA f16.
// M-tile = 32 rows/block: each wave reuses one B fragment for two WMMAs.
//
// Input order (jax pytree: dict keys sorted alphabetically):
//  0:x(100000x12) 1:edge_attr(600000x4) 2:edge_index(2x600000) 3:batch(100000)
//  4..9   edge_enc {b1,b2,beta,g,w1(4x128),w2(128x128)}
//  10     gate_b(1) 11 gate_w(128)
//  12..15 head {b1(128),b2(650),w1(128x128),w2(128x650)}
//  16+12i mp[i].edge{b1,b2,beta,g,w1(384x128),w2}, mp[i].node{b1,b2,beta,g,w1(256x128),w2}
//  52..57 node_enc {b1,b2,beta,g,w1(12x128),w2}
// ---------------------------------------------------------------------------

typedef _Float16 half_t;
typedef __attribute__((ext_vector_type(16))) _Float16 v16h;
typedef __attribute__((ext_vector_type(8)))  _Float16 v8h;
typedef __attribute__((ext_vector_type(8)))  float    v8f;

union V16H { v16h v; v8h h8[2]; };

#define HID 128
#define N_NODES 100000
#define N_EDGES 600000
#define N_GRAPHS 16
#define MIN_LOG_STD (-2.9957322735539909f)
#define MAX_LOG_STD (5.0f)
#define MROWS 32                     // rows per block (two 16-row WMMA tiles)

// --------------------------- weight convert --------------------------------
// W f32 [K][N] row-major  ->  WT f16 [N][Kpad] (zero padded along K)
__global__ void convert_w_T(const float* __restrict__ W, half_t* __restrict__ WT,
                            int K, int N, int Kpad) {
  int i = blockIdx.x * 256 + threadIdx.x;
  if (i >= N * Kpad) return;
  int n = i / Kpad, k = i % Kpad;
  WT[i] = (half_t)(k < K ? W[k * N + n] : 0.0f);
}

__global__ void zero_f32(float* __restrict__ p, long n) {
  long i = (long)blockIdx.x * 256 + threadIdx.x;
  if (i < n) p[i] = 0.0f;
}

__global__ void init_pool(unsigned* __restrict__ gmax_u, float* __restrict__ den,
                          float* __restrict__ pooled) {
  int t = threadIdx.x;
  if (t < N_GRAPHS) { gmax_u[t] = 0u; den[t] = 0.0f; }
  for (int i = t; i < N_GRAPHS * HID; i += 256) pooled[i] = 0.0f;
}

// --------------------------- fused MLP + LN --------------------------------
// MODE 0: A row = A0[row][0..in_w0) padded to K1 (=32)
// MODE 1: A row = [ h[s[row]] | h[r[row]] | e[row] ]   (K1=384), h=A0, e=A1
// MODE 2: A row = [ h[row] | agg[row] ]                (K1=256), h=A0, agg=A1
// out[row] = (resid ? resid[row] : 0) + LN(relu(A@W1+b1)@W2+b2)*g+beta
template <int K1, int MODE>
__global__ __launch_bounds__(256) void fused_mlp_ln(
    const float* __restrict__ A0, const float* __restrict__ A1,
    const int* __restrict__ s_idx, const int* __restrict__ r_idx,
    const half_t* __restrict__ W1T, const float* __restrict__ b1,
    const half_t* __restrict__ W2T, const float* __restrict__ b2,
    const float* __restrict__ g, const float* __restrict__ beta,
    const float* __restrict__ resid, float* __restrict__ out,
    int n_rows, int in_w0) {
  __shared__ __align__(16) half_t As[MROWS * K1];
  __shared__ __align__(16) half_t Hs[MROWS * HID];
  __shared__ float Ys[MROWS * HID];
  __shared__ float redS[MROWS * 8], redQ[MROWS * 8];
  __shared__ float muS[MROWS], rsS[MROWS];

  const int tid  = threadIdx.x;
  const int row0 = blockIdx.x * MROWS;

  // ---- stage A (f32 -> f16) into LDS, zero padded ----
  for (int idx = tid; idx < MROWS * K1; idx += 256) {
    int rr = idx / K1, k = idx - rr * K1;
    int row = row0 + rr;
    float v = 0.0f;
    if (row < n_rows) {
      if (MODE == 0) {
        if (k < in_w0) v = A0[row * in_w0 + k];
      } else if (MODE == 1) {
        if (k < HID)            v = A0[(long)s_idx[row] * HID + k];
        else if (k < 2 * HID)   v = A0[(long)r_idx[row] * HID + (k - HID)];
        else                    v = A1[(long)row * HID + (k - 2 * HID)];
      } else {  // MODE 2
        if (k < HID) v = A0[(long)row * HID + k];
        else         v = A1[(long)row * HID + (k - HID)];
      }
    }
    As[idx] = (half_t)v;
  }
  __syncthreads();

  const int wave  = tid >> 5;          // 0..7 -> output column tile
  const int lane  = tid & 31;
  const int lrow  = lane & 15;         // M (for A) / N (for B,C,D)
  const int khalf = (lane >= 16) ? 8 : 0;
  const int mbase = (lane >= 16) ? 8 : 0;
  const int ncol  = 16 * wave + lrow;

  // ---- GEMM1: hidden = relu(A @ W1 + b1), f16 into LDS ----
  v8f acc0 = {}, acc1 = {};
  for (int kk = 0; kk < K1; kk += 32) {
    V16H a0, a1, b;
    const half_t* bp = &W1T[(long)ncol * K1 + kk + khalf];
    __builtin_prefetch(bp + 32, 0, 1);                 // next K-chunk of W1T
    b.h8[0] = *(const v8h*)(bp);
    b.h8[1] = *(const v8h*)(bp + 16);
    const half_t* ap0 = &As[lrow * K1 + kk + khalf];
    const half_t* ap1 = &As[(16 + lrow) * K1 + kk + khalf];
    a0.h8[0] = *(const v8h*)(ap0);
    a0.h8[1] = *(const v8h*)(ap0 + 16);
    a1.h8[0] = *(const v8h*)(ap1);
    a1.h8[1] = *(const v8h*)(ap1 + 16);
    acc0 = __builtin_amdgcn_wmma_f32_16x16x32_f16(false, a0.v, false, b.v,
                                                  (short)0, acc0, false, false);
    acc1 = __builtin_amdgcn_wmma_f32_16x16x32_f16(false, a1.v, false, b.v,
                                                  (short)0, acc1, false, false);
  }
  {
    float bb = b1[ncol];
#pragma unroll
    for (int v = 0; v < 8; ++v) {
      float h0 = acc0[v] + bb, h1 = acc1[v] + bb;
      h0 = h0 > 0.0f ? h0 : 0.0f;
      h1 = h1 > 0.0f ? h1 : 0.0f;
      Hs[(mbase + v) * HID + ncol]      = (half_t)h0;
      Hs[(16 + mbase + v) * HID + ncol] = (half_t)h1;
    }
  }
  __syncthreads();

  // ---- GEMM2: y = hidden @ W2 + b2 ----
  v8f y0 = {}, y1 = {};
  for (int kk = 0; kk < HID; kk += 32) {
    V16H a0, a1, b;
    const half_t* bp = &W2T[(long)ncol * HID + kk + khalf];
    b.h8[0] = *(const v8h*)(bp);
    b.h8[1] = *(const v8h*)(bp + 16);
    const half_t* ap0 = &Hs[lrow * HID + kk + khalf];
    const half_t* ap1 = &Hs[(16 + lrow) * HID + kk + khalf];
    a0.h8[0] = *(const v8h*)(ap0);
    a0.h8[1] = *(const v8h*)(ap0 + 16);
    a1.h8[0] = *(const v8h*)(ap1);
    a1.h8[1] = *(const v8h*)(ap1 + 16);
    y0 = __builtin_amdgcn_wmma_f32_16x16x32_f16(false, a0.v, false, b.v,
                                                (short)0, y0, false, false);
    y1 = __builtin_amdgcn_wmma_f32_16x16x32_f16(false, a1.v, false, b.v,
                                                (short)0, y1, false, false);
  }
  {
    float bb = b2[ncol];
#pragma unroll
    for (int v = 0; v < 8; ++v) {
      Ys[(mbase + v) * HID + ncol]      = y0[v] + bb;
      Ys[(16 + mbase + v) * HID + ncol] = y1[v] + bb;
    }
  }
  __syncthreads();

  // ---- LayerNorm over 128 cols per row (32 rows x 8 partials of 16) ----
  {
    int rr = tid >> 3, part = tid & 7;
    const float* yr = &Ys[rr * HID + part * 16];
    float s = 0.0f, q = 0.0f;
#pragma unroll
    for (int i = 0; i < 16; ++i) { float y = yr[i]; s += y; q += y * y; }
    redS[rr * 8 + part] = s;
    redQ[rr * 8 + part] = q;
  }
  __syncthreads();
  if (tid < MROWS) {
    float s = 0.0f, q = 0.0f;
#pragma unroll
    for (int i = 0; i < 8; ++i) { s += redS[tid * 8 + i]; q += redQ[tid * 8 + i]; }
    float mu  = s * (1.0f / HID);
    float var = q * (1.0f / HID) - mu * mu;
    muS[tid] = mu;
    rsS[tid] = rsqrtf(var + 1e-5f);
  }
  __syncthreads();

  for (int idx = tid; idx < MROWS * HID; idx += 256) {
    int rr = idx >> 7, c = idx & 127;
    int row = row0 + rr;
    if (row < n_rows) {
      float y = (Ys[idx] - muS[rr]) * rsS[rr] * g[c] + beta[c];
      if (resid) y += resid[(long)row * HID + c];  // RES_SCALE = 1
      out[(long)row * HID + c] = y;
    }
  }
}

// --------------------------- graph scatter / pool --------------------------
__global__ void scatter_add_edges(const float* __restrict__ e,
                                  const int* __restrict__ r,
                                  float* __restrict__ agg, int n_edges) {
  long t = (long)blockIdx.x * 256 + threadIdx.x;
  if (t >= (long)n_edges * HID) return;
  int edge = (int)(t >> 7), c = (int)(t & 127);
  atomicAdd(&agg[(long)r[edge] * HID + c], e[t]);
}

__global__ void gate_kernel(const float* __restrict__ h, const float* __restrict__ gw,
                            const float* __restrict__ gb, const int* __restrict__ batch,
                            float* __restrict__ gate, unsigned* __restrict__ gmax_u,
                            int n_nodes) {
  int row  = blockIdx.x * 8 + (threadIdx.x >> 5);
  int lane = threadIdx.x & 31;
  if (row >= n_nodes) return;
  float s = 0.0f;
  for (int c = lane; c < HID; c += 32) s += h[(long)row * HID + c] * gw[c];
  for (int off = 16; off; off >>= 1) s += __shfl_down(s, off, 32);
  if (lane == 0) {
    s += gb[0];
    gate[row] = s;
    unsigned u   = __float_as_uint(s);
    unsigned key = (u & 0x80000000u) ? ~u : (u | 0x80000000u);  // monotone map
    atomicMax(&gmax_u[batch[row]], key);
  }
}

__global__ void pool_kernel(const float* __restrict__ h, const float* __restrict__ gate,
                            const int* __restrict__ batch, const unsigned* __restrict__ gmax_u,
                            float* __restrict__ den, float* __restrict__ pooled,
                            int n_nodes) {
  long t = (long)blockIdx.x * 256 + threadIdx.x;
  if (t >= (long)n_nodes * HID) return;
  int row = (int)(t >> 7), c = (int)(t & 127);
  int gph = batch[row];
  unsigned key = gmax_u[gph];
  float gmax = __uint_as_float((key & 0x80000000u) ? (key & 0x7fffffffu) : ~key);
  float wv = __expf(gate[row] - gmax);
  if (c == 0) atomicAdd(&den[gph], wv);
  atomicAdd(&pooled[gph * HID + c], wv * h[t]);
}

// --------------------------- head (tiny, one block) -------------------------
__global__ __launch_bounds__(256) void head_kernel(
    const float* __restrict__ pooled_raw, const float* __restrict__ den,
    const float* __restrict__ w1, const float* __restrict__ b1,
    const float* __restrict__ w2, const float* __restrict__ b2,
    float* __restrict__ out) {
  __shared__ float P[N_GRAPHS * HID];
  __shared__ float Hh[N_GRAPHS * HID];
  int tid = threadIdx.x;
  for (int i = tid; i < N_GRAPHS * HID; i += 256)
    P[i] = pooled_raw[i] / den[i >> 7];
  __syncthreads();
  for (int i = tid; i < N_GRAPHS * HID; i += 256) {
    int gph = i >> 7, c = i & 127;
    float s = b1[c];
    for (int k = 0; k < HID; ++k) s += P[gph * HID + k] * w1[k * HID + c];
    Hh[i] = s > 0.0f ? s : 0.0f;
  }
  __syncthreads();
  const int DOUT = 650;  // K_COMP * (1 + 2*Z_DIM)
  for (int i = tid; i < N_GRAPHS * DOUT; i += 256) {
    int gph = i / DOUT, c = i - gph * DOUT;
    float s = b2[c];
    for (int k = 0; k < HID; ++k) s += Hh[gph * HID + k] * w2[k * DOUT + c];
    int comp = c / 65, slot = c - comp * 65;
    if (slot == 0) {
      out[gph * 10 + comp] = s;                                    // logits
    } else if (slot <= 32) {
      out[160 + (gph * 10 + comp) * 32 + (slot - 1)] = s;          // mu
    } else {
      float ls = fminf(fmaxf(s, MIN_LOG_STD), MAX_LOG_STD);
      out[160 + 5120 + (gph * 10 + comp) * 32 + (slot - 33)] = ls; // log_std
    }
  }
}

// ---------------------------------------------------------------------------
extern "C" void kernel_launch(void* const* d_in, const int* in_sizes, int n_in,
                              void* d_out, int out_size, void* d_ws, size_t ws_size,
                              hipStream_t stream) {
  (void)in_sizes; (void)n_in; (void)out_size; (void)ws_size;
  const float* x         = (const float*)d_in[0];
  const float* edge_attr = (const float*)d_in[1];
  const int*   s_idx     = (const int*)d_in[2];
  const int*   r_idx     = s_idx + N_EDGES;
  const int*   batch     = (const int*)d_in[3];

  // params (alphabetical pytree order)
  const float* ee_b1 = (const float*)d_in[4];
  const float* ee_b2 = (const float*)d_in[5];
  const float* ee_bt = (const float*)d_in[6];
  const float* ee_g  = (const float*)d_in[7];
  const float* ee_w1 = (const float*)d_in[8];
  const float* ee_w2 = (const float*)d_in[9];
  const float* gate_b = (const float*)d_in[10];
  const float* gate_w = (const float*)d_in[11];
  const float* hd_b1 = (const float*)d_in[12];
  const float* hd_b2 = (const float*)d_in[13];
  const float* hd_w1 = (const float*)d_in[14];
  const float* hd_w2 = (const float*)d_in[15];
  const float* ne_b1 = (const float*)d_in[52];
  const float* ne_b2 = (const float*)d_in[53];
  const float* ne_bt = (const float*)d_in[54];
  const float* ne_g  = (const float*)d_in[55];
  const float* ne_w1 = (const float*)d_in[56];
  const float* ne_w2 = (const float*)d_in[57];

  // ---- workspace carve (256B aligned) ----
  char* ws = (char*)d_ws;
  size_t off = 0;
  auto carve = [&](size_t bytes) -> void* {
    void* p = ws + off;
    off = (off + bytes + 255) & ~(size_t)255;
    return p;
  };
  float* h    = (float*)carve((size_t)N_NODES * HID * 4);
  float* e    = (float*)carve((size_t)N_EDGES * HID * 4);
  float* agg  = (float*)carve((size_t)N_NODES * HID * 4);
  float* gate = (float*)carve((size_t)N_NODES * 4);
  unsigned* gmax_u = (unsigned*)carve(N_GRAPHS * 4);
  float* den    = (float*)carve(N_GRAPHS * 4);
  float* pooled = (float*)carve((size_t)N_GRAPHS * HID * 4);
  half_t* ne_w1T = (half_t*)carve((size_t)HID * 32 * 2);
  half_t* ne_w2T = (half_t*)carve((size_t)HID * HID * 2);
  half_t* ee_w1T = (half_t*)carve((size_t)HID * 32 * 2);
  half_t* ee_w2T = (half_t*)carve((size_t)HID * HID * 2);
  half_t* mp_ew1T[3], *mp_ew2T[3], *mp_nw1T[3], *mp_nw2T[3];
  for (int i = 0; i < 3; ++i) {
    mp_ew1T[i] = (half_t*)carve((size_t)HID * 384 * 2);
    mp_ew2T[i] = (half_t*)carve((size_t)HID * HID * 2);
    mp_nw1T[i] = (half_t*)carve((size_t)HID * 256 * 2);
    mp_nw2T[i] = (half_t*)carve((size_t)HID * HID * 2);
  }

  auto cvt = [&](const float* W, half_t* WT, int K, int N, int Kpad) {
    int n = N * Kpad;
    convert_w_T<<<(n + 255) / 256, 256, 0, stream>>>(W, WT, K, N, Kpad);
  };
  cvt(ne_w1, ne_w1T, 12, HID, 32);
  cvt(ne_w2, ne_w2T, HID, HID, HID);
  cvt(ee_w1, ee_w1T, 4, HID, 32);
  cvt(ee_w2, ee_w2T, HID, HID, HID);
  for (int i = 0; i < 3; ++i) {
    const float* ew1 = (const float*)d_in[16 + 12 * i + 4];
    const float* ew2 = (const float*)d_in[16 + 12 * i + 5];
    const float* nw1 = (const float*)d_in[16 + 12 * i + 10];
    const float* nw2 = (const float*)d_in[16 + 12 * i + 11];
    cvt(ew1, mp_ew1T[i], 384, HID, 384);
    cvt(ew2, mp_ew2T[i], HID, HID, HID);
    cvt(nw1, mp_nw1T[i], 256, HID, 256);
    cvt(nw2, mp_nw2T[i], HID, HID, HID);
  }

  const int nb_node = (N_NODES + MROWS - 1) / MROWS;
  const int nb_edge = (N_EDGES + MROWS - 1) / MROWS;

  // encoders
  fused_mlp_ln<32, 0><<<nb_node, 256, 0, stream>>>(
      x, nullptr, nullptr, nullptr, ne_w1T, ne_b1, ne_w2T, ne_b2, ne_g, ne_bt,
      nullptr, h, N_NODES, 12);
  fused_mlp_ln<32, 0><<<nb_edge, 256, 0, stream>>>(
      edge_attr, nullptr, nullptr, nullptr, ee_w1T, ee_b1, ee_w2T, ee_b2, ee_g, ee_bt,
      nullptr, e, N_EDGES, 4);

  // message passing
  for (int i = 0; i < 3; ++i) {
    const float* eb1 = (const float*)d_in[16 + 12 * i + 0];
    const float* eb2 = (const float*)d_in[16 + 12 * i + 1];
    const float* ebt = (const float*)d_in[16 + 12 * i + 2];
    const float* eg  = (const float*)d_in[16 + 12 * i + 3];
    const float* nb1 = (const float*)d_in[16 + 12 * i + 6];
    const float* nb2 = (const float*)d_in[16 + 12 * i + 7];
    const float* nbt = (const float*)d_in[16 + 12 * i + 8];
    const float* ng  = (const float*)d_in[16 + 12 * i + 9];

    // e = e + LN(mlp([h[s]|h[r]|e]))  (in place: each block owns its rows)
    fused_mlp_ln<384, 1><<<nb_edge, 256, 0, stream>>>(
        h, e, s_idx, r_idx, mp_ew1T[i], eb1, mp_ew2T[i], eb2, eg, ebt,
        e, e, N_EDGES, 0);

    // agg = segment_sum(e, r)
    long nagg = (long)N_NODES * HID;
    zero_f32<<<(unsigned)((nagg + 255) / 256), 256, 0, stream>>>(agg, nagg);
    long nsc = (long)N_EDGES * HID;
    scatter_add_edges<<<(unsigned)((nsc + 255) / 256), 256, 0, stream>>>(e, r_idx, agg, N_EDGES);

    // h = h + LN(mlp([h|agg]))
    fused_mlp_ln<256, 2><<<nb_node, 256, 0, stream>>>(
        h, agg, nullptr, nullptr, mp_nw1T[i], nb1, mp_nw2T[i], nb2, ng, nbt,
        h, h, N_NODES, 0);
  }

  // attention pooling
  init_pool<<<1, 256, 0, stream>>>(gmax_u, den, pooled);
  gate_kernel<<<(N_NODES + 7) / 8, 256, 0, stream>>>(h, gate_w, gate_b, batch, gate,
                                                     gmax_u, N_NODES);
  long npool = (long)N_NODES * HID;
  pool_kernel<<<(unsigned)((npool + 255) / 256), 256, 0, stream>>>(
      h, gate, batch, gmax_u, den, pooled, N_NODES);

  // head
  head_kernel<<<1, 256, 0, stream>>>(pooled, den, hd_w1, hd_b1, hd_w2, hd_b2,
                                     (float*)d_out);
}
